// PS_Cell_68719477375
// MI455X (gfx1250) — matrix operand order repaired and checked by reading, not compiled, not run on hardware
//
#include <hip/hip_runtime.h>

#define NN 65536   // total nodes
#define BB 64      // graphs
#define NPG 1024   // nodes per graph
#define EE 1048576 // edges
#define DD 128     // feature dim
#define KTOP 512   // kept nodes per graph

typedef __attribute__((ext_vector_type(16))) __bf16 v16bf;
typedef __attribute__((ext_vector_type(8)))  float  v8f;

// ---------------- degree / normalization ----------------

__global__ void deg_init_kernel(float* __restrict__ deg) {
    int i = blockIdx.x * blockDim.x + threadIdx.x;
    deg[i] = 1.0f;  // self-loop weight
}

__global__ void deg_accum_kernel(const int* __restrict__ dst,
                                 const float* __restrict__ w,
                                 float* __restrict__ deg) {
    int e = blockIdx.x * blockDim.x + threadIdx.x;
    atomicAdd(&deg[dst[e]], w[e]);
}

__global__ void deg_inv_kernel(float* __restrict__ deg) {
    int i = blockIdx.x * blockDim.x + threadIdx.x;
    float d = deg[i];
    deg[i] = (d > 0.f) ? rsqrtf(fmaxf(d, 1e-12f)) : 0.f;
}

__global__ void pnorm_kernel(const float* __restrict__ p, float* __restrict__ out) {
    __shared__ float s[DD];
    int t = threadIdx.x;
    float v = p[t];
    s[t] = v * v;
    __syncthreads();
    for (int off = DD / 2; off > 0; off >>= 1) {
        if (t < off) s[t] += s[t + off];
        __syncthreads();
    }
    if (t == 0) out[0] = rsqrtf(fmaxf(s[0], 1e-30f));
}

// ---------------- h = x @ W via WMMA (bf16 in, f32 acc) ----------------
// Block: 256 threads = 8 waves; each wave owns 16 rows x all 128 cols.
// W transposed into LDS as bf16 with row stride 144 (32B-aligned, de-conflicted).

__global__ void __launch_bounds__(256)
gemm_h_kernel(const float* __restrict__ x, const float* __restrict__ W,
              float* __restrict__ h) {
    __shared__ __bf16 WT[DD * 144];   // WT[n*144 + k] = W[k][n]

    const int tid = threadIdx.x;
    for (int i = tid; i < DD * DD; i += 256) {
        int n = i >> 7, k = i & 127;
        WT[n * 144 + k] = (__bf16)W[k * DD + n];
    }
    __syncthreads();

    const int wave    = tid >> 5;
    const int lane    = tid & 31;
    const int halfsel = (lane < 16) ? 0 : 1;
    const int lrow    = lane & 15;
    const int m0      = blockIdx.x * 128 + wave * 16;
    const int arow    = m0 + lrow;

    // Preload the 4 A fragments covering K = 0..127.
    v16bf afrag[4];
#pragma unroll
    for (int kk = 0; kk < 4; ++kk) {
        const int k0 = kk * 32;
        const float4* xp = (const float4*)(x + (size_t)arow * DD + k0 + halfsel * 16);
        float fl[16];
        ((float4*)fl)[0] = xp[0];   // K = kbase+0..3
        ((float4*)fl)[1] = xp[1];   // K = kbase+4..7
        ((float4*)fl)[2] = xp[2];   // K = kbase+8..11
        ((float4*)fl)[3] = xp[3];   // K = kbase+12..15
#pragma unroll
        for (int i = 0; i < 16; ++i) afrag[kk][i] = (__bf16)fl[i];
    }

#pragma unroll
    for (int n0 = 0; n0 < 8; ++n0) {
        v8f acc = {};
#pragma unroll
        for (int kk = 0; kk < 4; ++kk) {
            const int k0 = kk * 32;
            const __bf16* bp = &WT[(n0 * 16 + lrow) * 144 + k0 + halfsel * 16];
            v16bf bfrag = *(const v16bf*)bp;
            acc = __builtin_amdgcn_wmma_f32_16x16x32_bf16(
                false, afrag[kk], false, bfrag, (short)0, acc, false, false);
        }
        float* hp = h + (size_t)(m0 + halfsel * 8) * DD + n0 * 16 + lrow;
#pragma unroll
        for (int r = 0; r < 8; ++r) hp[(size_t)r * DD] = acc[r];
    }
}

// ---------------- aggregation: self-loops init then edge scatter ----------------

__global__ void aggr_init_kernel(const float* __restrict__ h,
                                 const float* __restrict__ dinv,
                                 float* __restrict__ aggr) {
    int t = blockIdx.x * blockDim.x + threadIdx.x;
    int n = t >> 5, c = t & 31;
    float s = dinv[n];
    s = s * s;  // self-loop norm: dinv[n] * 1 * dinv[n]
    float4 v = ((const float4*)(h + (size_t)n * DD))[c];
    v.x *= s; v.y *= s; v.z *= s; v.w *= s;
    ((float4*)(aggr + (size_t)n * DD))[c] = v;
}

__global__ void __launch_bounds__(256)
edge_scatter_kernel(const int* __restrict__ src, const int* __restrict__ dst,
                    const float* __restrict__ w, const float* __restrict__ dinv,
                    const float* __restrict__ h, float* __restrict__ aggr) {
    int wid = threadIdx.x >> 5, lane = threadIdx.x & 31;
    int e = blockIdx.x * 8 + wid;
    int s = src[e], d = dst[e];
    float nrm = dinv[s] * w[e] * dinv[d];
    float4 hv = ((const float4*)(h + (size_t)s * DD))[lane];
    float* ap = aggr + (size_t)d * DD + lane * 4;
    atomicAdd(ap + 0, nrm * hv.x);
    atomicAdd(ap + 1, nrm * hv.y);
    atomicAdd(ap + 2, nrm * hv.z);
    atomicAdd(ap + 3, nrm * hv.w);
}

// ---------------- bias + relu + projection score (in place) ----------------

__global__ void __launch_bounds__(256)
finalize_score_kernel(float* __restrict__ aggr, const float* __restrict__ b,
                      const float* __restrict__ p, const float* __restrict__ ipn,
                      float* __restrict__ score) {
    int wid = threadIdx.x >> 5, lane = threadIdx.x & 31;
    int n = blockIdx.x * 8 + wid;
    float4 v  = ((float4*)(aggr + (size_t)n * DD))[lane];
    float4 bv = ((const float4*)b)[lane];
    float4 pv = ((const float4*)p)[lane];
    v.x = fmaxf(v.x + bv.x, 0.f);
    v.y = fmaxf(v.y + bv.y, 0.f);
    v.z = fmaxf(v.z + bv.z, 0.f);
    v.w = fmaxf(v.w + bv.w, 0.f);
    ((float4*)(aggr + (size_t)n * DD))[lane] = v;
    float dot = v.x * pv.x + v.y * pv.y + v.z * pv.z + v.w * pv.w;
#pragma unroll
    for (int off = 16; off > 0; off >>= 1) dot += __shfl_down(dot, off, 32);
    if (lane == 0) score[n] = dot * ipn[0];
}

// ---------------- per-graph top-k (bitonic) + tanh-gated mean ----------------

__global__ void __launch_bounds__(NPG)
topk_pool_kernel(const float* __restrict__ hfin, const float* __restrict__ score,
                 float* __restrict__ out) {
    __shared__ float key[NPG];
    __shared__ int   sidx[NPG];
    __shared__ float acc[DD];
    const int t = threadIdx.x;
    const int g = blockIdx.x;

    key[t]  = score[g * NPG + t];
    sidx[t] = t;
    if (t < DD) acc[t] = 0.f;
    __syncthreads();

    // bitonic sort, descending by score
    for (int k = 2; k <= NPG; k <<= 1) {
        for (int j = k >> 1; j > 0; j >>= 1) {
            int ixj = t ^ j;
            if (ixj > t) {
                bool descRegion = ((t & k) == 0);
                float a = key[t], c = key[ixj];
                bool doSwap = descRegion ? (a < c) : (a > c);
                if (doSwap) {
                    key[t] = c; key[ixj] = a;
                    int ia = sidx[t]; sidx[t] = sidx[ixj]; sidx[ixj] = ia;
                }
            }
            __syncthreads();
        }
    }

    // gated mean over the kept KTOP nodes
    const int grp = t >> 7, c = t & 127;
    float accp = 0.f;
    for (int kk = grp; kk < KTOP; kk += 8) {
        float gate = tanhf(key[kk]);
        int node = g * NPG + sidx[kk];
        accp += gate * hfin[(size_t)node * DD + c];
    }
    atomicAdd(&acc[c], accp);
    __syncthreads();
    if (t < DD) out[g * DD + t] = acc[t] * (1.0f / (float)KTOP);
}

// ---------------- launcher ----------------

extern "C" void kernel_launch(void* const* d_in, const int* in_sizes, int n_in,
                              void* d_out, int out_size, void* d_ws, size_t ws_size,
                              hipStream_t stream) {
    (void)in_sizes; (void)n_in; (void)out_size; (void)ws_size;
    const float* x  = (const float*)d_in[0];
    const float* ew = (const float*)d_in[1];
    const float* W  = (const float*)d_in[2];
    const float* b  = (const float*)d_in[3];
    const float* p  = (const float*)d_in[4];
    const int*   ei = (const int*)d_in[5];   // [2, E]: row0 = src, row1 = dst
    const int* src = ei;
    const int* dst = ei + EE;

    float* ws    = (float*)d_ws;
    float* dinv  = ws;                        // N floats (deg -> dinv in place)
    float* h     = dinv + NN;                 // N*D floats
    float* aggr  = h + (size_t)NN * DD;       // N*D floats (becomes relu'd h)
    float* score = aggr + (size_t)NN * DD;    // N floats
    float* ipn   = score + NN;                // 1 float

    float* out = (float*)d_out;               // [B, D]

    deg_init_kernel  <<<NN / 256, 256, 0, stream>>>(dinv);
    deg_accum_kernel <<<EE / 256, 256, 0, stream>>>(dst, ew, dinv);
    deg_inv_kernel   <<<NN / 256, 256, 0, stream>>>(dinv);
    pnorm_kernel     <<<1, DD, 0, stream>>>(p, ipn);
    gemm_h_kernel    <<<NN / 128, 256, 0, stream>>>(x, W, h);
    aggr_init_kernel <<<(NN * 32) / 256, 256, 0, stream>>>(h, dinv, aggr);
    edge_scatter_kernel  <<<EE / 8, 256, 0, stream>>>(src, dst, ew, dinv, h, aggr);
    finalize_score_kernel<<<NN / 8, 256, 0, stream>>>(aggr, b, p, ipn, score);
    topk_pool_kernel <<<BB, NPG, 0, stream>>>(aggr, score, out);
}
